// Grok1Attention_850403524959
// MI455X (gfx1250) — matrix operand order repaired and checked by reading, not compiled, not run on hardware
//
#include <hip/hip_runtime.h>
#include <hip/hip_bf16.h>

typedef __bf16 bf16_t;
typedef __attribute__((ext_vector_type(16))) __bf16 v16bf;
typedef __attribute__((ext_vector_type(8)))  __bf16 v8bf;
typedef __attribute__((ext_vector_type(8)))  float  v8f;

#define T_SEQ   2048
#define D_MODEL 6144
#define NH      48
#define NKV     8
#define HD      128
#define QKV_N   8192            // (NH + 2*NKV) * HD
#define GQA_G   (NH / NKV)      // 6

// ---------------------------------------------------------------------------
// WMMA helpers (CDNA5 wave32, v_wmma_f32_16x16x32_bf16)
// ---------------------------------------------------------------------------
__device__ __forceinline__ v8f wmma_bf16(v16bf a, v16bf b, v8f c) {
  return __builtin_amdgcn_wmma_f32_16x16x32_bf16(
      /*neg_a=*/false, a, /*neg_b=*/false, b,
      /*c_mod=*/(short)0, c, /*reuse_a=*/false, /*reuse_b=*/false);
}

// A-matrix 16x32 bf16 fragment (ISA 7.12.2): lane = M row; lane<16 holds
// K=0..7 (v0-3) and K=16..23 (v4-7); lane>=16 holds K=8..15 and K=24..31.
// `base` points at element [row0][k0] of a row-major array with `stride`.
__device__ __forceinline__ v16bf load_fragA(const bf16_t* base, int stride) {
  int lane = threadIdx.x & 31;
  int m    = lane & 15;
  int hs   = (lane >> 4) * 8;
  const bf16_t* p = base + (size_t)m * stride + hs;
  v8bf lo = *reinterpret_cast<const v8bf*>(p);
  v8bf hi = *reinterpret_cast<const v8bf*>(p + 16);
  return __builtin_shufflevector(lo, hi, 0,1,2,3,4,5,6,7,8,9,10,11,12,13,14,15);
}

// B-matrix 32x16 bf16 fragment: lane = N column; lane<16 holds K=0..7 and
// K=16..23, lane>=16 holds K=8..15 and K=24..31 (same K order as fragA).
// `base` points at element [n0][k0] of an N-major array with `stride`.
__device__ __forceinline__ v16bf load_fragB(const bf16_t* base, int stride) {
  int lane = threadIdx.x & 31;
  int n    = lane & 15;
  int kh   = (lane >> 4) * 8;
  const bf16_t* p = base + (size_t)n * stride + kh;
  v8bf lo = *reinterpret_cast<const v8bf*>(p);
  v8bf hi = *reinterpret_cast<const v8bf*>(p + 16);
  return __builtin_shufflevector(lo, hi, 0,1,2,3,4,5,6,7,8,9,10,11,12,13,14,15);
}

// ---------------------------------------------------------------------------
// f32 -> bf16 conversion (memory bound)
// ---------------------------------------------------------------------------
__global__ void cvt_f32_bf16(const float* __restrict__ in,
                             bf16_t* __restrict__ out, size_t n) {
  size_t i = (size_t)blockIdx.x * blockDim.x + threadIdx.x;
  size_t stride = (size_t)gridDim.x * blockDim.x;
  for (; i < n; i += stride) out[i] = (bf16_t)in[i];
}

// ---------------------------------------------------------------------------
// Tiled bf16 WMMA GEMM: C[M,N] f32 = A[M,K] bf16 @ B[K,N] bf16
// BM=128 BN=128 BK=32, 256 threads = 8 waves (4x2), 32x64 per wave.
// Double-buffered LDS + software pipeline: one barrier per K-step, global
// loads for tile k+1 issued before the WMMA burst on tile k.
// ---------------------------------------------------------------------------
#define BM 128
#define BN 128
#define BK 32
#define LDSP (BK + 8)   // padded K stride: 40 bf16 = 80B (16B-aligned rows)

__global__ __launch_bounds__(256) void gemm_bf16_wmma(
    const bf16_t* __restrict__ A, const bf16_t* __restrict__ B,
    float* __restrict__ C, int M, int N, int K) {
  __shared__ __align__(16) bf16_t As[2][BM][LDSP];
  __shared__ __align__(16) bf16_t Bt[2][BN][LDSP];  // B transposed (N-major)

  const int bm  = blockIdx.y * BM;
  const int bn  = blockIdx.x * BN;
  const int tid = threadIdx.x;
  const int w   = tid >> 5;
  const int wm  = (w >> 1) * 32;       // 0,32,64,96
  const int wn  = (w & 1) * 64;        // 0,64

  v8f acc[2][4] = {};

  const int ar = tid >> 1;             // A stage: row 0..127
  const int ac = (tid & 1) * 16;       // A stage: col 0 or 16
  const int bk = tid >> 3;             // B stage: k row 0..31
  const int bc = (tid & 7) * 16;       // B stage: n col chunk

  // ---- prologue: fetch tile 0 into registers, commit to buffer 0 ----
  v8bf a_r0, a_r1, b_r0, b_r1;
  {
    const bf16_t* asrc = &A[(size_t)(bm + ar) * K + ac];
    a_r0 = *reinterpret_cast<const v8bf*>(asrc);
    a_r1 = *reinterpret_cast<const v8bf*>(asrc + 8);
    const bf16_t* bsrc = &B[(size_t)bk * N + bn + bc];
    b_r0 = *reinterpret_cast<const v8bf*>(bsrc);
    b_r1 = *reinterpret_cast<const v8bf*>(bsrc + 8);
  }
  *reinterpret_cast<v8bf*>(&As[0][ar][ac])     = a_r0;
  *reinterpret_cast<v8bf*>(&As[0][ar][ac + 8]) = a_r1;
#pragma unroll
  for (int i = 0; i < 8; ++i) Bt[0][bc + i][bk]     = b_r0[i];
#pragma unroll
  for (int i = 0; i < 8; ++i) Bt[0][bc + 8 + i][bk] = b_r1[i];

  int cur = 0;
  for (int k0 = 0; k0 < K; k0 += BK) {
    __syncthreads();                       // buf[cur] ready; buf[cur^1] free
    const bool more = (k0 + BK) < K;
    if (more) {                            // fetch next tile (overlaps WMMA)
      const bf16_t* asrc = &A[(size_t)(bm + ar) * K + k0 + BK + ac];
      a_r0 = *reinterpret_cast<const v8bf*>(asrc);
      a_r1 = *reinterpret_cast<const v8bf*>(asrc + 8);
      const bf16_t* bsrc = &B[(size_t)(k0 + BK + bk) * N + bn + bc];
      b_r0 = *reinterpret_cast<const v8bf*>(bsrc);
      b_r1 = *reinterpret_cast<const v8bf*>(bsrc + 8);
      if (k0 + 2 * BK < K) {               // prefetch tile k+2 -> global_prefetch_b8
        __builtin_prefetch(&A[(size_t)(bm + ar) * K + k0 + 2 * BK + ac], 0, 1);
        __builtin_prefetch(&B[(size_t)(k0 + 2 * BK + bk) * N + bn + bc], 0, 1);
      }
    }

    // ---- WMMA burst on buf[cur]: 6 fragment loads feed 8 WMMAs ----
    v16bf a0 = load_fragA(&As[cur][wm][0],      LDSP);
    v16bf a1 = load_fragA(&As[cur][wm + 16][0], LDSP);
    v16bf bf[4];
#pragma unroll
    for (int j = 0; j < 4; ++j)
      bf[j] = load_fragB(&Bt[cur][wn + j * 16][0], LDSP);
#pragma unroll
    for (int j = 0; j < 4; ++j) {
      acc[0][j] = wmma_bf16(a0, bf[j], acc[0][j]);
      acc[1][j] = wmma_bf16(a1, bf[j], acc[1][j]);
    }

    if (more) {                            // commit next tile to buf[cur^1]
      *reinterpret_cast<v8bf*>(&As[cur ^ 1][ar][ac])     = a_r0;
      *reinterpret_cast<v8bf*>(&As[cur ^ 1][ar][ac + 8]) = a_r1;
#pragma unroll
      for (int i = 0; i < 8; ++i) Bt[cur ^ 1][bc + i][bk]     = b_r0[i];
#pragma unroll
      for (int i = 0; i < 8; ++i) Bt[cur ^ 1][bc + 8 + i][bk] = b_r1[i];
    }
    cur ^= 1;
  }

  // epilogue: C/D layout (ISA 7.12.2): lane = N col (&15), lane half selects M+8
  const int lane = tid & 31;
  const int nn   = lane & 15;
  const int mh   = (lane >> 4) * 8;
#pragma unroll
  for (int i = 0; i < 2; ++i)
#pragma unroll
    for (int j = 0; j < 4; ++j) {
      const int rbase = bm + wm + i * 16 + mh;
      const int cbase = bn + wn + j * 16 + nn;
#pragma unroll
      for (int e = 0; e < 8; ++e)
        C[(size_t)(rbase + e) * N + cbase] = acc[i][j][e];
    }
}

// ---------------------------------------------------------------------------
// RoPE (NeoX) on Q,K; V straight convert. qkv f32 [T, 8192] -> bf16 Q/K/V.
// ---------------------------------------------------------------------------
__global__ __launch_bounds__(256) void rope_kernel(
    const int* __restrict__ positions, const float* __restrict__ qkv,
    bf16_t* __restrict__ q, bf16_t* __restrict__ k, bf16_t* __restrict__ v) {
  const int t = blockIdx.x;
  const float pos = (float)positions[t];
  const float* row = qkv + (size_t)t * QKV_N;

  // Q: NH heads x 64 rotation pairs
  for (int idx = threadIdx.x; idx < NH * 64; idx += blockDim.x) {
    int h = idx >> 6, i = idx & 63;
    float inv = powf(10000.0f, -(float)(2 * i) / 128.0f);
    float f = pos * inv;
    float c = cosf(f), s = sinf(f);
    float x1 = row[h * HD + i];
    float x2 = row[h * HD + i + 64];
    bf16_t* o = q + ((size_t)t * NH + h) * HD;
    o[i]      = (bf16_t)(x1 * c - x2 * s);
    o[i + 64] = (bf16_t)(x2 * c + x1 * s);
  }
  // K: NKV heads x 64 rotation pairs (offset NH*HD = 6144)
  for (int idx = threadIdx.x; idx < NKV * 64; idx += blockDim.x) {
    int h = idx >> 6, i = idx & 63;
    float inv = powf(10000.0f, -(float)(2 * i) / 128.0f);
    float f = pos * inv;
    float c = cosf(f), s = sinf(f);
    float x1 = row[NH * HD + h * HD + i];
    float x2 = row[NH * HD + h * HD + i + 64];
    bf16_t* o = k + ((size_t)t * NKV + h) * HD;
    o[i]      = (bf16_t)(x1 * c - x2 * s);
    o[i + 64] = (bf16_t)(x2 * c + x1 * s);
  }
  // V: straight convert (offset (NH+NKV)*HD = 7168)
  for (int idx = threadIdx.x; idx < NKV * HD; idx += blockDim.x)
    v[(size_t)t * NKV * HD + idx] = (bf16_t)row[(NH + NKV) * HD + idx];
}

// ---------------------------------------------------------------------------
// Causal GQA attention, tanh-capped logits, flash-style accumulation.
// Grid (T/64, NH), 128 threads = 4 waves; wave w owns q-rows [q0+16w, +16).
// No running max needed: |logit| <= 30 -> exp sums fit easily in f32.
// ---------------------------------------------------------------------------
__global__ __launch_bounds__(128) void attn_kernel(
    const bf16_t* __restrict__ Q, const bf16_t* __restrict__ Kg,
    const bf16_t* __restrict__ Vg, bf16_t* __restrict__ Og) {
  __shared__ __align__(16) bf16_t Ks[64][HD + 8];    // keys row-major (hd contiguous)
  __shared__ __align__(16) bf16_t Vt[HD][64 + 8];    // V transposed (hd-major)
  __shared__ __align__(16) bf16_t Ps[4][16][64];     // per-wave P round-trip

  const int qb  = blockIdx.x;
  const int h   = blockIdx.y;
  const int kvh = h / GQA_G;
  const int tid = threadIdx.x;
  const int w   = tid >> 5;
  const int lane = tid & 31;
  const int q0  = qb * 64;
  const int wq  = q0 + w * 16;          // wave's query-row base
  const int mh  = (lane >> 4) * 8;
  const int nn  = lane & 15;
  const float scale = 0.08838834764831845f;  // 1/sqrt(128)

  // Q fragments for this wave's 16 rows, all of HD (4 k-steps of 32)
  v16bf qf[4];
#pragma unroll
  for (int ks = 0; ks < 4; ++ks)
    qf[ks] = load_fragA(&Q[((size_t)wq * NH + h) * HD + ks * 32], NH * HD);

  v8f   o[8]  = {};
  float den[8] = {0.f, 0.f, 0.f, 0.f, 0.f, 0.f, 0.f, 0.f};

  for (int kb = 0; kb <= qb; ++kb) {
    const int s0 = kb * 64;
    __syncthreads();
    // stage K row-major and V transposed: 2 threads per key row
    {
      const int r  = tid >> 1;
      const int c0 = (tid & 1) * 64;
      const bf16_t* ksrc = &Kg[((size_t)(s0 + r) * NKV + kvh) * HD + c0];
      const bf16_t* vsrc = &Vg[((size_t)(s0 + r) * NKV + kvh) * HD + c0];
#pragma unroll
      for (int i = 0; i < 8; ++i)
        *reinterpret_cast<v8bf*>(&Ks[r][c0 + i * 8]) =
            *reinterpret_cast<const v8bf*>(ksrc + i * 8);
#pragma unroll
      for (int i = 0; i < 8; ++i) {
        v8bf x = *reinterpret_cast<const v8bf*>(vsrc + i * 8);
#pragma unroll
        for (int e = 0; e < 8; ++e) Vt[c0 + i * 8 + e][r] = x[e];
      }
    }
    __syncthreads();

    // S = Q K^T: 4 key tiles of 16; then cap/mask/exp -> P (bf16 in LDS)
#pragma unroll
    for (int j = 0; j < 4; ++j) {
      v8f s = {};
#pragma unroll
      for (int ks = 0; ks < 4; ++ks) {
        v16bf b = load_fragB(&Ks[j * 16][ks * 32], HD + 8);
        s = wmma_bf16(qf[ks], b, s);
      }
#pragma unroll
      for (int e = 0; e < 8; ++e) {
        const int qrow = wq + mh + e;
        const int key  = s0 + j * 16 + nn;
        float l = s[e] * scale;
        l = 30.0f * tanhf(l * (1.0f / 30.0f));
        float p = (key <= qrow) ? __expf(l) : 0.0f;
        den[e] += p;
        Ps[w][mh + e][j * 16 + nn] = (bf16_t)p;
      }
    }

    // O += P V : P is per-wave in LDS (in-wave DS ordering via dscnt waits)
#pragma unroll
    for (int ks = 0; ks < 2; ++ks) {
      v16bf a = load_fragA(&Ps[w][0][ks * 32], 64);
#pragma unroll
      for (int j = 0; j < 8; ++j) {
        v16bf b = load_fragB(&Vt[j * 16][ks * 32], 64 + 8);
        o[j] = wmma_bf16(a, b, o[j]);
      }
    }
  }

  // reduce den across the 16 lanes of each half (wave32-safe shuffles)
#pragma unroll
  for (int e = 0; e < 8; ++e) {
    float d = den[e];
    d += __shfl_xor(d, 1, 32);
    d += __shfl_xor(d, 2, 32);
    d += __shfl_xor(d, 4, 32);
    d += __shfl_xor(d, 8, 32);
    den[e] = d;
  }

  // normalize and write attn output bf16 [T, NH*HD]
#pragma unroll
  for (int j = 0; j < 8; ++j)
#pragma unroll
    for (int e = 0; e < 8; ++e) {
      float val = o[j][e] / den[e];
      Og[(size_t)(wq + mh + e) * (NH * HD) + h * HD + j * 16 + nn] = (bf16_t)val;
    }
}

// ---------------------------------------------------------------------------
// Host-side orchestration
// ---------------------------------------------------------------------------
extern "C" void kernel_launch(void* const* d_in, const int* in_sizes, int n_in,
                              void* d_out, int out_size, void* d_ws, size_t ws_size,
                              hipStream_t stream) {
  const int*   positions = (const int*)  d_in[0];
  const float* hidden    = (const float*)d_in[1];
  const float* w_qkv     = (const float*)d_in[2];
  const float* w_o       = (const float*)d_in[3];

  char* ws = (char*)d_ws;
  size_t off = 0;
  bf16_t* hid_b  = (bf16_t*)(ws + off); off += (size_t)T_SEQ * D_MODEL * 2;   // 25.2 MB
  bf16_t* wqkv_b = (bf16_t*)(ws + off); off += (size_t)D_MODEL * QKV_N * 2;   // 100.7 MB
  bf16_t* wo_b   = (bf16_t*)(ws + off); off += (size_t)D_MODEL * D_MODEL * 2; // 75.5 MB
  float*  qkv_f  = (float*) (ws + off); off += (size_t)T_SEQ * QKV_N * 4;     // 67 MB
  bf16_t* q_b    = (bf16_t*)(ws + off); off += (size_t)T_SEQ * NH  * HD * 2;  // 25.2 MB
  bf16_t* k_b    = (bf16_t*)(ws + off); off += (size_t)T_SEQ * NKV * HD * 2;  // 4.2 MB
  bf16_t* v_b    = (bf16_t*)(ws + off); off += (size_t)T_SEQ * NKV * HD * 2;  // 4.2 MB
  bf16_t* attn_b = (bf16_t*)(ws + off); off += (size_t)T_SEQ * D_MODEL * 2;   // 25.2 MB

  // 1) one-time bf16 conversion of activations + weights
  cvt_f32_bf16<<<2048, 256, 0, stream>>>(hidden, hid_b,  (size_t)T_SEQ * D_MODEL);
  cvt_f32_bf16<<<4096, 256, 0, stream>>>(w_qkv,  wqkv_b, (size_t)D_MODEL * QKV_N);
  cvt_f32_bf16<<<4096, 256, 0, stream>>>(w_o,    wo_b,   (size_t)D_MODEL * D_MODEL);

  // 2) QKV projection: [2048,6144] @ [6144,8192] -> f32
  dim3 g1(QKV_N / BN, T_SEQ / BM);   // (64, 16)
  gemm_bf16_wmma<<<g1, 256, 0, stream>>>(hid_b, wqkv_b, qkv_f,
                                         T_SEQ, QKV_N, D_MODEL);

  // 3) RoPE + split to bf16 Q/K/V
  rope_kernel<<<T_SEQ, 256, 0, stream>>>(positions, qkv_f, q_b, k_b, v_b);

  // 4) causal GQA attention with tanh cap
  dim3 g2(T_SEQ / 64, NH);           // (32, 48)
  attn_kernel<<<g2, 128, 0, stream>>>(q_b, k_b, v_b, attn_b);

  // 5) output projection: [2048,6144] @ [6144,6144] -> f32 d_out
  dim3 g3(D_MODEL / BN, T_SEQ / BM); // (48, 16)
  gemm_bf16_wmma<<<g3, 256, 0, stream>>>(attn_b, wo_b, (float*)d_out,
                                         T_SEQ, D_MODEL, D_MODEL);
}
